// YOLOLoss_63247688401213
// MI455X (gfx1250) — compile-verified, matrix operand order checked by reading, not run on hardware
//
#include <hip/hip_runtime.h>
#include <hip/hip_bf16.h>
#include <math.h>

typedef __attribute__((ext_vector_type(16))) _Float16 v16h;
typedef __attribute__((ext_vector_type(8)))  float    v8f;

#define BN 32      // batch
#define AN 8400    // anchors (80*80 + 40*40 + 20*20)
#define GN 60      // max GT
#define GP 64      // GT padded to 4 WMMA tiles
#define CP 96      // classes padded to 3 K-chunks of 32
#define NC 80
#define TK 10
#define ATILES 525 // 8400 / 16 (exact)

static constexpr size_t NBA = (size_t)BN * AN;
constexpr size_t al256(size_t x){ return (x + 255) & ~(size_t)255; }
constexpr size_t OFF_BOXES = 0;
constexpr size_t OFF_BSUM  = al256(OFF_BOXES + NBA*4*sizeof(float));
constexpr size_t OFF_FGANY = al256(OFF_BSUM  + NBA*sizeof(float));
constexpr size_t OFF_BCE16 = al256(OFF_FGANY + NBA*sizeof(float));
constexpr size_t OFF_COST  = al256(OFF_BCE16 + NBA*CP*sizeof(_Float16));
constexpr size_t OFF_IOU   = al256(OFF_COST  + (size_t)BN*GP*AN*sizeof(float));
constexpr size_t OFF_CNT   = al256(OFF_IOU   + (size_t)BN*GP*AN*sizeof(float));
constexpr size_t OFF_SELG  = al256(OFF_CNT   + NBA*sizeof(int));
constexpr size_t OFF_MATCH = al256(OFF_SELG  + NBA*sizeof(int));
constexpr size_t OFF_PIOU  = al256(OFF_MATCH + NBA*sizeof(int));
constexpr size_t OFF_FGF   = al256(OFF_PIOU  + NBA*sizeof(float));
constexpr size_t OFF_ACC   = al256(OFF_FGF   + NBA*sizeof(int));

__device__ __forceinline__ float sigmoidf_(float x){ return 1.0f/(1.0f + __expf(-x)); }
__device__ __forceinline__ float bcelogit(float x, float t){
  return fmaxf(x, 0.0f) - x*t + log1pf(__expf(-fabsf(x)));
}

struct AInfo { const float* base; int cs; int ix, iy; float cx, cy, st; };

__device__ __forceinline__ AInfo anchor_info(const float* f8, const float* f16,
                                             const float* f32, int b, int a){
  AInfo r;
  if (a < 6400){
    int y = a/80, x = a - y*80;
    r.cs = 6400; r.base = f8  + (size_t)b*85*6400 + (size_t)y*80 + x;
    r.ix = x; r.iy = y; r.st = 8.f;  r.cx = (x+0.5f)*8.f;  r.cy = (y+0.5f)*8.f;
  } else if (a < 8000){
    int t = a-6400; int y = t/40, x = t - y*40;
    r.cs = 1600; r.base = f16 + (size_t)b*85*1600 + (size_t)y*40 + x;
    r.ix = x; r.iy = y; r.st = 16.f; r.cx = (x+0.5f)*16.f; r.cy = (y+0.5f)*16.f;
  } else {
    int t = a-8000; int y = t/20, x = t - y*20;
    r.cs = 400;  r.base = f32 + (size_t)b*85*400  + (size_t)y*20 + x;
    r.ix = x; r.iy = y; r.st = 32.f; r.cx = (x+0.5f)*32.f; r.cy = (y+0.5f)*32.f;
  }
  return r;
}

// ---------------- init ----------------
__global__ void k_init(int* cnt, int* selg, float* acc){
  int i = blockIdx.x*blockDim.x + threadIdx.x;
  if (i < (int)NBA){ cnt[i] = 0; selg[i] = 0; }
  if (i < 4) acc[i] = 0.f;
}

// ------- decode + per-anchor BCE terms + fg-any mask -------
__global__ __launch_bounds__(256) void k_decode(
    const float* f8, const float* f16, const float* f32,
    const float* gtb, const int* ngts,
    float* boxes, float* bsum, float* fgany, _Float16* bce16){
  int idx = blockIdx.x*blockDim.x + threadIdx.x;
  if (idx >= (int)NBA) return;
  int b = idx / AN, a = idx - b*AN;
  AInfo ai = anchor_info(f8, f16, f32, b, a);

  float tx = ai.base[0], ty = ai.base[ai.cs];
  float tw = ai.base[2*(size_t)ai.cs], th = ai.base[3*(size_t)ai.cs];
  float to = ai.base[4*(size_t)ai.cs];
  float4 bb;
  bb.x = (tx + ai.ix)*ai.st; bb.y = (ty + ai.iy)*ai.st;
  bb.z = __expf(tw)*ai.st;   bb.w = __expf(th)*ai.st;
  *(float4*)(boxes + (size_t)idx*4) = bb;

  float so = sigmoidf_(to);
  float s0 = 0.f;
  _Float16* bp = bce16 + (size_t)idx*CP;
  #pragma unroll 4
  for (int c = 0; c < NC; c++){
    float xl = ai.base[(size_t)(5+c)*ai.cs];
    float p  = sqrtf(sigmoidf_(xl)*so);
    p = fminf(fmaxf(p, 1e-7f), 1.0f - 1e-7f);
    float b1 = -__logf(p), b0 = -__logf(1.0f - p);
    s0 += b0;
    bp[c] = (_Float16)(b1 - b0);
  }
  #pragma unroll
  for (int c = NC; c < CP; c++) bp[c] = (_Float16)0.0f;
  bsum[idx] = s0;

  int ng = ngts[b];
  bool fg = false;
  for (int g = 0; g < GN; g++){
    if (g >= ng) break;
    const float* gp = gtb + ((size_t)b*GN + g)*4;
    float gx = gp[0], gy = gp[1], gw = gp[2], gh = gp[3];
    bool ib = ai.cx > gx-0.5f*gw && ai.cx < gx+0.5f*gw &&
              ai.cy > gy-0.5f*gh && ai.cy < gy+0.5f*gh;
    float R = 2.5f*ai.st;
    bool ic = ai.cx > gx-R && ai.cx < gx+R && ai.cy > gy-R && ai.cy < gy+R;
    fg = fg || ib || ic;
  }
  fgany[idx] = fg ? 1.0f : 0.0f;
}

// ------- WMMA cost kernel: onehot(gtc) x diff^T fused with IoU/penalties -------
__global__ __launch_bounds__(256) void k_cost(
    const _Float16* bce16, const float* bsum, const float* fgany,
    const float* boxes, const float* gtb, const int* gtc, const int* ngts,
    float* cost, float* iouM){
  const int lane  = threadIdx.x & 31;
  const int wave  = threadIdx.x >> 5;
  const int atile = blockIdx.x*8 + wave;
  if (atile >= ATILES) return;            // uniform per wave -> EXEC all-1s for WMMA
  const int gtile = blockIdx.y;
  const int b     = blockIdx.z;
  const int khalf = lane >> 4;
  const int l15   = lane & 15;

  // A fragment: one-hot rows of GT classes (zero padding rows). 16x32 f16 layout:
  // lanes 0-15 M=lane, lanes 16-31 same M but K-halves offset by 8.
  int g0  = gtile*16 + l15;
  int cls = (g0 < GN) ? gtc[b*GN + g0] : -1;
  v16h afrag[3];
  #pragma unroll
  for (int kc = 0; kc < 3; kc++){
    #pragma unroll
    for (int e = 0; e < 8; e++){
      int ka = kc*32 + khalf*8 + e;
      int kb = ka + 16;
      afrag[kc][e]     = (ka == cls) ? (_Float16)1.0f : (_Float16)0.0f;
      afrag[kc][8 + e] = (kb == cls) ? (_Float16)1.0f : (_Float16)0.0f;
    }
  }

  const int a = atile*16 + l15;
  const _Float16* bp = bce16 + ((size_t)b*AN + a)*CP;
  v8f acc = {0.f,0.f,0.f,0.f,0.f,0.f,0.f,0.f};
  #pragma unroll
  for (int kc = 0; kc < 3; kc++){
    v16h bfrag;
    const _Float16* s0p = bp + kc*32 + khalf*8;   // 16B-aligned contiguous run
    const _Float16* s1p = s0p + 16;               // second run
    #pragma unroll
    for (int e = 0; e < 8; e++){ bfrag[e] = s0p[e]; bfrag[8+e] = s1p[e]; }
    acc = __builtin_amdgcn_wmma_f32_16x16x32_f16(false, afrag[kc], false, bfrag,
                                                 (short)0, acc, false, false);
  }

  // epilogue: fuse bce0_sum + 3*(-log iou) + geometric / validity penalties
  float4 pb  = *(const float4*)(boxes + ((size_t)b*AN + a)*4);
  float  bs  = bsum[(size_t)b*AN + a];
  bool   fgA = fgany[(size_t)b*AN + a] > 0.5f;
  float cx, cy, st;
  if (a < 6400){ int y=a/80, x=a-y*80;          st=8.f;  cx=(x+0.5f)*8.f;  cy=(y+0.5f)*8.f;  }
  else if (a < 8000){ int t=a-6400, y=t/40, x=t-y*40; st=16.f; cx=(x+0.5f)*16.f; cy=(y+0.5f)*16.f; }
  else { int t=a-8000, y=t/20, x=t-y*20;        st=32.f; cx=(x+0.5f)*32.f; cy=(y+0.5f)*32.f; }
  int ng = ngts[b];

  #pragma unroll
  for (int j = 0; j < 8; j++){
    int g = gtile*16 + khalf*8 + j;   // C-layout: lane group selects M+8, VGPR j = M
    if (g >= GN) continue;
    const float* gp = gtb + ((size_t)b*GN + g)*4;
    float gx = gp[0], gy = gp[1], gw = gp[2], gh = gp[3];
    float tlx = fmaxf(gx-0.5f*gw, pb.x-0.5f*pb.z);
    float tly = fmaxf(gy-0.5f*gh, pb.y-0.5f*pb.w);
    float brx = fminf(gx+0.5f*gw, pb.x+0.5f*pb.z);
    float bry = fminf(gy+0.5f*gh, pb.y+0.5f*pb.w);
    float inter = (tlx < brx && tly < bry) ? (brx-tlx)*(bry-tly) : 0.f;
    float iou = inter / (gw*gh + pb.z*pb.w - inter + 1e-16f);
    bool ib = cx > gx-0.5f*gw && cx < gx+0.5f*gw && cy > gy-0.5f*gh && cy < gy+0.5f*gh;
    float Rr = 2.5f*st;
    bool ic = cx > gx-Rr && cx < gx+Rr && cy > gy-Rr && cy < gy+Rr;
    bool inboth = ib && ic;
    bool valid  = g < ng;
    float cv = acc[j] + bs - 3.0f*__logf(iou + 1e-8f)
             + (inboth ? 0.f : 100000.f)
             + (fgA    ? 0.f : 1.0e9f)
             + (valid  ? 0.f : 1.0e9f);
    size_t o = ((size_t)b*GP + g)*AN + a;
    cost[o] = cv;
    iouM[o] = iou;
  }
}

// ------- per-(b,g): dyn_k from top-10 IoU, then bottom-10 cost selection -------
__global__ __launch_bounds__(256) void k_assign(
    const float* iouM, const float* cost, const float* fgany, const int* ngts,
    int* cnt, int* selg){
  const int g   = blockIdx.x;
  const int b   = blockIdx.y;
  const int tid = threadIdx.x;
  __shared__ float sv[256*TK];
  __shared__ int   si[256*TK];
  size_t row = ((size_t)b*GP + g)*AN;
  const float* fga = fgany + (size_t)b*AN;

  float lv[TK]; int li[TK];
  // phase 1: top-10 of iou*fg -> dyn_k
  #pragma unroll
  for (int k = 0; k < TK; k++){ lv[k] = -1.0f; li[k] = 0; }
  for (int a = tid; a < AN; a += 256){
    float v = iouM[row + a] * fga[a];
    if (v > lv[TK-1]){
      int p = TK-1;
      while (p > 0 && lv[p-1] < v){ lv[p]=lv[p-1]; li[p]=li[p-1]; p--; }
      lv[p] = v; li[p] = a;
    }
  }
  #pragma unroll
  for (int k = 0; k < TK; k++){ sv[tid*TK+k]=lv[k]; si[tid*TK+k]=li[k]; }
  __syncthreads();
  int dynk = 1;
  if (tid == 0){
    float bv[TK];
    #pragma unroll
    for (int k = 0; k < TK; k++) bv[k] = -1.0f;
    for (int i = 0; i < 256*TK; i++){
      float v = sv[i];
      if (v > bv[TK-1]){
        int p = TK-1;
        while (p > 0 && bv[p-1] < v){ bv[p]=bv[p-1]; p--; }
        bv[p] = v;
      }
    }
    float s = 0.f;
    #pragma unroll
    for (int k = 0; k < TK; k++) s += fmaxf(bv[k], 0.f);
    int dk = (int)s; if (dk < 1) dk = 1;
    dynk = dk;
  }
  __syncthreads();

  // phase 2: bottom-10 of cost (as top-10 of -cost)
  #pragma unroll
  for (int k = 0; k < TK; k++){ lv[k] = -3.0e38f; li[k] = 0; }
  for (int a = tid; a < AN; a += 256){
    float v = -cost[row + a];
    if (v > lv[TK-1]){
      int p = TK-1;
      while (p > 0 && lv[p-1] < v){ lv[p]=lv[p-1]; li[p]=li[p-1]; p--; }
      lv[p] = v; li[p] = a;
    }
  }
  #pragma unroll
  for (int k = 0; k < TK; k++){ sv[tid*TK+k]=lv[k]; si[tid*TK+k]=li[k]; }
  __syncthreads();
  if (tid == 0 && g < ngts[b]){
    float bv[TK]; int bi[TK];
    #pragma unroll
    for (int k = 0; k < TK; k++){ bv[k] = -3.0e38f; bi[k] = 0; }
    for (int i = 0; i < 256*TK; i++){
      float v = sv[i]; int id = si[i];
      if (v > bv[TK-1]){
        int p = TK-1;
        while (p > 0 && bv[p-1] < v){ bv[p]=bv[p-1]; bi[p]=bi[p-1]; p--; }
        bv[p] = v; bi[p] = id;
      }
    }
    for (int k = 0; k < TK && k < dynk; k++){
      int a = bi[k];
      atomicAdd(&cnt[(size_t)b*AN + a], 1);
      atomicExch(&selg[(size_t)b*AN + a], g);
    }
  }
}

// ------- per-anchor conflict resolution -------
__global__ void k_resolve(const int* cnt, const int* selg, const float* cost,
                          const float* iouM, int* fgf, int* match, float* piou){
  int idx = blockIdx.x*blockDim.x + threadIdx.x;
  if (idx >= (int)NBA) return;
  int b = idx / AN, a = idx - b*AN;
  int c = cnt[idx];
  int g = 0; float pi = 0.f; int fg = 0;
  if (c == 1){ g = selg[idx]; fg = 1; }
  else if (c > 1){
    float bc = 3.0e38f;
    for (int gg = 0; gg < GN; gg++){
      float cc = cost[((size_t)b*GP + gg)*AN + a];
      if (cc < bc){ bc = cc; g = gg; }
    }
    fg = 1;
  }
  if (fg) pi = iouM[((size_t)b*GP + g)*AN + a];
  fgf[idx] = fg; match[idx] = g; piou[idx] = pi;
}

// ------- fused losses with block reduction -------
__global__ __launch_bounds__(256) void k_loss(
    const float* f8, const float* f16, const float* f32,
    const float* boxes, const float* gtb, const int* gtc,
    const int* fgf, const int* match, const float* piou, float* acc){
  int idx = blockIdx.x*blockDim.x + threadIdx.x;
  int tid = threadIdx.x;
  float vfg = 0.f, viou = 0.f, vobj = 0.f, vcls = 0.f;
  if (idx < (int)NBA){
    int b = idx / AN, a = idx - b*AN;
    AInfo ai = anchor_info(f8, f16, f32, b, a);
    float obj = ai.base[4*(size_t)ai.cs];
    int fg = fgf[idx];
    vfg  = (float)fg;
    vobj = bcelogit(obj, fg ? 0.9f : 0.0f);
    if (fg){
      int g = match[idx];
      const float* gp = gtb + ((size_t)b*GN + g)*4;
      float4 pbv = *(const float4*)(boxes + (size_t)idx*4);
      float tlx = fmaxf(pbv.x-0.5f*pbv.z, gp[0]-0.5f*gp[2]);
      float tly = fmaxf(pbv.y-0.5f*pbv.w, gp[1]-0.5f*gp[3]);
      float brx = fminf(pbv.x+0.5f*pbv.z, gp[0]+0.5f*gp[2]);
      float bry = fminf(pbv.y+0.5f*pbv.w, gp[1]+0.5f*gp[3]);
      float inter = (tlx < brx && tly < bry) ? (brx-tlx)*(bry-tly) : 0.f;
      float uni = pbv.z*pbv.w + gp[2]*gp[3] - inter;
      float iou = inter / (uni + 1e-16f);
      viou = 1.0f - iou*iou;
      int   cg = gtc[b*GN + g];
      float pi = piou[idx];
      for (int c = 0; c < NC; c++){
        float xl = ai.base[(size_t)(5+c)*ai.cs];
        vcls += bcelogit(xl, (c == cg) ? pi : 0.0f);
      }
    }
  }
  __shared__ float red[256];
  float sums[4] = {vfg, viou, vobj, vcls};
  for (int s = 0; s < 4; s++){
    red[tid] = sums[s]; __syncthreads();
    for (int off = 128; off > 0; off >>= 1){
      if (tid < off) red[tid] += red[tid + off];
      __syncthreads();
    }
    if (tid == 0) atomicAdd(&acc[s], red[0]);
    __syncthreads();
  }
}

__global__ void k_final(const float* acc, float* out){
  if (threadIdx.x == 0 && blockIdx.x == 0){
    float nfg = fmaxf(acc[0], 1.0f);
    out[0] = (5.0f*acc[1] + acc[2] + acc[3]) / nfg;
  }
}

extern "C" void kernel_launch(void* const* d_in, const int* in_sizes, int n_in,
                              void* d_out, int out_size, void* d_ws, size_t ws_size,
                              hipStream_t stream) {
  (void)in_sizes; (void)n_in; (void)out_size; (void)ws_size;
  const float* f8  = (const float*)d_in[0];
  const float* f16 = (const float*)d_in[1];
  const float* f32 = (const float*)d_in[2];
  const float* gtb = (const float*)d_in[3];
  const int*   gtc = (const int*)d_in[4];
  const int*   ngt = (const int*)d_in[5];
  float* out = (float*)d_out;

  char* ws = (char*)d_ws;
  float*     boxes = (float*)(ws + OFF_BOXES);
  float*     bsum  = (float*)(ws + OFF_BSUM);
  float*     fgany = (float*)(ws + OFF_FGANY);
  _Float16*  bce16 = (_Float16*)(ws + OFF_BCE16);
  float*     cost  = (float*)(ws + OFF_COST);
  float*     iouM  = (float*)(ws + OFF_IOU);
  int*       cnt   = (int*)(ws + OFF_CNT);
  int*       selg  = (int*)(ws + OFF_SELG);
  int*       match = (int*)(ws + OFF_MATCH);
  float*     piou  = (float*)(ws + OFF_PIOU);
  int*       fgf   = (int*)(ws + OFF_FGF);
  float*     acc   = (float*)(ws + OFF_ACC);

  const int T = 256;
  const int NB_BLK = (int)((NBA + T - 1) / T);   // 1050

  k_init   <<<NB_BLK, T, 0, stream>>>(cnt, selg, acc);
  k_decode <<<NB_BLK, T, 0, stream>>>(f8, f16, f32, gtb, ngt,
                                      boxes, bsum, fgany, bce16);
  k_cost   <<<dim3((ATILES + 7)/8, GP/16, BN), T, 0, stream>>>(
              bce16, bsum, fgany, boxes, gtb, gtc, ngt, cost, iouM);
  k_assign <<<dim3(GN, BN), T, 0, stream>>>(iouM, cost, fgany, ngt, cnt, selg);
  k_resolve<<<NB_BLK, T, 0, stream>>>(cnt, selg, cost, iouM, fgf, match, piou);
  k_loss   <<<NB_BLK, T, 0, stream>>>(f8, f16, f32, boxes, gtb, gtc,
                                      fgf, match, piou, acc);
  k_final  <<<1, 32, 0, stream>>>(acc, out);
}
